// PointFeaturePyramid_76570676953179
// MI455X (gfx1250) — compile-verified, matrix operand order checked by reading, not run on hardware
//
#include <hip/hip_runtime.h>
#include <hip/hip_bf16.h>
#include <math.h>

typedef __attribute__((ext_vector_type(16))) _Float16 v16h;
typedef __attribute__((ext_vector_type(8)))  float    v8f;

#define BATCH 2048
#define NRAW  360
#define KNN_K 8
#define DSF   4
#define KQ    90     // downsampled points per row
#define KQP   96     // padded to multiple of 32 for WMMA K
#define D1    256
#define D2    128
#define D3    128

// ---------------------------------------------------------------------------
// KNN + downsample: for each (b, q=0..89), mean of the 8 values with smallest
// |x[b,4q] - x[b,j]| over j=0..359. Row cached in LDS, per-thread top-8 kept
// as a register-resident sorted (ascending) array; insertion is a predicated
// one-pass bubble (branch taken only when candidate beats current max).
// ---------------------------------------------------------------------------
__global__ void knn_ds_kernel(const float* __restrict__ x, _Float16* __restrict__ h0) {
  __shared__ float sx[NRAW];
  const int b   = blockIdx.x;
  const int tid = threadIdx.x;
  for (int i = tid; i < NRAW; i += blockDim.x) sx[i] = x[(size_t)b * NRAW + i];
  __syncthreads();

  if (tid < KQP) {
    if (tid < KQ) {
      const float xi = sx[tid * DSF];
      float d[KNN_K], v[KNN_K];
#pragma unroll
      for (int j = 0; j < KNN_K; ++j) { d[j] = 3.0e38f; v[j] = 0.0f; }
      for (int j = 0; j < NRAW; ++j) {
        const float val  = sx[j];
        const float dist = fabsf(val - xi);
        if (dist < d[KNN_K - 1]) {           // strict '<' => earliest index wins ties
          d[KNN_K - 1] = dist; v[KNN_K - 1] = val;
#pragma unroll
          for (int t = KNN_K - 1; t > 0; --t) {
            const bool  sw = d[t] < d[t - 1];
            const float dt = d[t], dp = d[t - 1];
            const float vt = v[t], vp = v[t - 1];
            d[t] = sw ? dp : dt; d[t - 1] = sw ? dt : dp;
            v[t] = sw ? vp : vt; v[t - 1] = sw ? vt : vp;
          }
        }
      }
      float s = 0.0f;
#pragma unroll
      for (int j = 0; j < KNN_K; ++j) s += v[j];
      h0[(size_t)b * KQP + tid] = (_Float16)(s * (1.0f / KNN_K));
    } else {
      h0[(size_t)b * KQP + tid] = (_Float16)0.0f;   // zero K-padding
    }
  }
}

// ---------------------------------------------------------------------------
// Pack f32 row-major weights (K x N) into f16 WMMA B-fragment-major layout:
//   dst[((nt*KSTEPS + ks)*32 + lane)*16 + h]
// where (per CDNA5 ISA 7.12.2, 16-bit B 32x16):
//   r = h/2, p = h%2, q = (r>=4)*2 + (lane>=16)
//   K = ks*32 + q*8 + 2*(r&3) + p,  N = nt*16 + (lane&15)
// Rows >= `rows` (K padding) are zero-filled. Each lane's fragment is then a
// single contiguous 32-byte chunk -> two global_load_b128 in the GEMM.
// ---------------------------------------------------------------------------
__global__ void pack_b_kernel(const float* __restrict__ src, _Float16* __restrict__ dst,
                              int rows, int cols, int kpad) {
  const int idx = blockIdx.x * blockDim.x + threadIdx.x;
  if (idx >= kpad * cols) return;
  const int h      = idx & 15;
  const int lane   = (idx >> 4) & 31;
  const int t      = idx >> 9;            // nt * ksteps + ks
  const int ksteps = kpad >> 5;
  const int nt = t / ksteps, ks = t - nt * ksteps;
  const int r = h >> 1, p = h & 1;
  const int q = ((r >> 2) << 1) | (lane >> 4);
  const int K = ks * 32 + q * 8 + ((r & 3) << 1) + p;
  const int N = nt * 16 + (lane & 15);
  dst[idx] = (K < rows) ? (_Float16)src[(size_t)K * cols + N] : (_Float16)0.0f;
}

// ---------------------------------------------------------------------------
// WMMA GEMM, fully unrolled K loop (KSTEPS steps of K=32), 4 waves/block each
// owning a 16-row M tile, NT=2 N-tiles per wave sharing one A fragment.
// A: M x K row-major f16 (lda).  Bf: fragment-packed (see pack_b_kernel).
// D = relu(A*B + bias), optional tanh; stored f16 (OUTH) or f32.
// ---------------------------------------------------------------------------
template <int KSTEPS, int ACT, bool OUTH>
__global__ __launch_bounds__(128) void wmma_gemm_kernel(
    const _Float16* __restrict__ A, int lda,
    const v16h* __restrict__ Bf,
    const float* __restrict__ bias,
    _Float16* __restrict__ outh, float* __restrict__ outf, int ldo) {
  constexpr int NT = 2;
  const int wave = threadIdx.x >> 5;
  const int lane = threadIdx.x & 31;
  const int mt = blockIdx.x * 4 + wave;
  const int ntBase = blockIdx.y * NT;
  const int l15 = lane & 15;
  const int hi  = lane >> 4;

  // Fold the lane-half K offset (q bit 0) into the row pointer: chunks at
  // k0 + 0..7 and k0 + 16..23 (relative), both 16B contiguous.
  const _Float16* Arow = A + (size_t)(mt * 16 + l15) * lda + hi * 8;

  v8f c[NT];
#pragma unroll
  for (int n = 0; n < NT; ++n) c[n] = {};

#pragma unroll
  for (int ks = 0; ks < KSTEPS; ++ks) {
    const int k0 = ks * 32;
    v16h a;
#pragma unroll
    for (int r = 0; r < 8; ++r) {
      const int kk = k0 + ((r >> 2) << 4) + ((r & 3) << 1);
      a[2 * r]     = Arow[kk];
      a[2 * r + 1] = Arow[kk + 1];
    }
#pragma unroll
    for (int n = 0; n < NT; ++n) {
      const v16h bfrag = Bf[(size_t)((ntBase + n) * KSTEPS + ks) * 32 + lane];
      c[n] = __builtin_amdgcn_wmma_f32_16x16x32_f16(
          /*neg_a=*/false, a, /*neg_b=*/false, bfrag,
          /*c_mod=*/(short)0, c[n], /*reuse_a=*/false, /*reuse_b=*/false);
    }
  }

#pragma unroll
  for (int n = 0; n < NT; ++n) {
    const int   ncol = (ntBase + n) * 16 + l15;
    const float bv   = bias[ncol];
#pragma unroll
    for (int r = 0; r < 8; ++r) {
      const int mrow = mt * 16 + r + hi * 8;
      float val = c[n][r] + bv;
      val = fmaxf(val, 0.0f);
      if (ACT) val = tanhf(val);
      if (OUTH) outh[(size_t)mrow * ldo + ncol] = (_Float16)val;
      else      outf[(size_t)mrow * ldo + ncol] = val;
    }
  }
}

// ---------------------------------------------------------------------------
extern "C" void kernel_launch(void* const* d_in, const int* in_sizes, int n_in,
                              void* d_out, int out_size, void* d_ws, size_t ws_size,
                              hipStream_t stream) {
  const float* x  = (const float*)d_in[0];
  const float* W1 = (const float*)d_in[1];
  const float* b1 = (const float*)d_in[2];
  const float* W2 = (const float*)d_in[3];
  const float* b2 = (const float*)d_in[4];
  const float* W3 = (const float*)d_in[5];
  const float* b3 = (const float*)d_in[6];
  float* out = (float*)d_out;

  char* ws = (char*)d_ws;
  size_t off = 0;
  _Float16* H0  = (_Float16*)(ws + off); off += (size_t)BATCH * KQP * 2;   // A of GEMM1
  _Float16* W1f = (_Float16*)(ws + off); off += (size_t)KQP * D1 * 2;      // packed B frags
  _Float16* W2f = (_Float16*)(ws + off); off += (size_t)D1 * D2 * 2;
  _Float16* W3f = (_Float16*)(ws + off); off += (size_t)D2 * D3 * 2;
  _Float16* H1  = (_Float16*)(ws + off); off += (size_t)BATCH * D1 * 2;
  _Float16* H2  = (_Float16*)(ws + off); off += (size_t)BATCH * D2 * 2;

  // Stage 1: KNN + downsample + f16 pack (one block per batch row)
  knn_ds_kernel<<<BATCH, 128, 0, stream>>>(x, H0);

  // Stage 2: weight conversion + WMMA-fragment packing
  pack_b_kernel<<<(KQP * D1 + 255) / 256, 256, 0, stream>>>(W1, W1f, KQ, D1, KQP);
  pack_b_kernel<<<(D1  * D2 + 255) / 256, 256, 0, stream>>>(W2, W2f, D1, D2, D1);
  pack_b_kernel<<<(D2  * D3 + 255) / 256, 256, 0, stream>>>(W3, W3f, D2, D3, D2);

  // Stage 3: WMMA MLP with fused epilogues
  wmma_gemm_kernel<KQP / 32, 0, true><<<dim3(BATCH / 64, D1 / 32), 128, 0, stream>>>(
      H0, KQP, (const v16h*)W1f, b1, H1, nullptr, D1);
  wmma_gemm_kernel<D1 / 32, 0, true><<<dim3(BATCH / 64, D2 / 32), 128, 0, stream>>>(
      H1, D1, (const v16h*)W2f, b2, H2, nullptr, D2);
  wmma_gemm_kernel<D2 / 32, 1, false><<<dim3(BATCH / 64, D3 / 32), 128, 0, stream>>>(
      H2, D2, (const v16h*)W3f, b3, nullptr, out, D3);
}
